// PFNTransformerLayer_56521769616166
// MI455X (gfx1250) — matrix operand order repaired and checked by reading, not compiled, not run on hardware
//
#include <hip/hip_runtime.h>

// ---------------------------------------------------------------------------
// CDNA5 (gfx1250) wave32 WMMA bf16 implementation of the PFN transformer layer
// ---------------------------------------------------------------------------

typedef __attribute__((ext_vector_type(16))) __bf16 bfvec16;
typedef __attribute__((ext_vector_type(8)))  float  fvec8;

union Frag16 { uint4 u[2]; bfvec16 v; };

__device__ __forceinline__ unsigned short f2bf(float f) {
  unsigned int u = __float_as_uint(f);
  u += 0x7fffu + ((u >> 16) & 1u);           // round-to-nearest-even
  return (unsigned short)(u >> 16);
}

// A fragment (16x32 bf16): lane row = lane%16; half-wave selects K sub-block.
// VGPR0..3 = K[half*8 .. half*8+8), VGPR4..7 = K[16+half*8 .. +8)  (ISA 7.12.2)
__device__ __forceinline__ bfvec16 frag_a(const unsigned short* p, int half) {
  Frag16 f;
  f.u[0] = *(const uint4*)(p + half * 8);
  f.u[1] = *(const uint4*)(p + half * 8 + 16);
  return f.v;
}

// B fragment (32x16 bf16): lane col = lane%16; K = [half*8 .. +8) then [16+half*8 .. +8)
// contiguous when B is stored K-contiguous per column (transposed weights).
__device__ __forceinline__ bfvec16 frag_b(const unsigned short* p, int half) {
  Frag16 f;
  f.u[0] = *(const uint4*)(p + half * 8);
  f.u[1] = *(const uint4*)(p + half * 8 + 16);
  return f.v;
}

__device__ __forceinline__ fvec8 wmma_bf16(bfvec16 a, bfvec16 b, fvec8 c) {
  return __builtin_amdgcn_wmma_f32_16x16x32_bf16(false, a, false, b,
                                                 (short)0, c, false, false);
}

// ---------------------------------------------------------------------------
// f32 -> bf16 conversion kernels
// ---------------------------------------------------------------------------
__global__ void cvt_bf16_kernel(const float* __restrict__ in,
                                unsigned short* __restrict__ out, long n) {
  long i = (long)blockIdx.x * blockDim.x + threadIdx.x;
  if (i < n) out[i] = f2bf(in[i]);
}

// dst[e][c*R + r] = bf16(src[e][r*C + c])  (batched transpose: weights -> [N,K])
__global__ void cvt_transpose_kernel(const float* __restrict__ in,
                                     unsigned short* __restrict__ out,
                                     int Eb, int R, int C) {
  long total = (long)Eb * R * C;
  long i = (long)blockIdx.x * blockDim.x + threadIdx.x;
  if (i >= total) return;
  long rc = (long)R * C;
  long e = i / rc, rem = i % rc;
  long r = rem / C, c = rem % C;
  out[e * rc + c * R + r] = f2bf(in[i]);
}

__global__ void zero_kernel(float* __restrict__ p, int n) {
  if ((int)threadIdx.x < n) p[threadIdx.x] = 0.f;
}

// ---------------------------------------------------------------------------
// Generic WMMA GEMM: C[M,N] = epilogue(A[M,K](bf16) x BT[N,K](bf16))
// Each wave computes a 16x64 tile; block = 4 waves -> 64x64 tile.
// Epilogue options: +bias, relu, per-row scale (MoE comb), +resid, accumulate,
// f32 and/or bf16 output, bf16 scatter mode 1 = V^T layout [(b*N+n)*S + s].
// ---------------------------------------------------------------------------
__global__ __launch_bounds__(128) void gemm_bf16_wmma(
    const unsigned short* __restrict__ A, const unsigned short* __restrict__ BT,
    const float* __restrict__ bias, float* __restrict__ Cf,
    unsigned short* __restrict__ Cb, const float* __restrict__ resid,
    const float* __restrict__ rowscale, int rs_stride, int rs_off,
    int M, int N, int K, int relu, int accum, int bf_mode, int batchB) {
  const int lane = threadIdx.x & 31;
  const int wave = threadIdx.x >> 5;
  const int half = lane >> 4;
  const int lcol = lane & 15;
  const long m0 = (long)blockIdx.x * 64 + wave * 16;
  const long n0 = (long)blockIdx.y * 64;

  fvec8 c[4] = {};
  const unsigned short* aRow = A + (m0 + lcol) * (long)K;
  for (int k0 = 0; k0 < K; k0 += 32) {
    bfvec16 a = frag_a(aRow + k0, half);
#pragma unroll
    for (int t = 0; t < 4; ++t) {
      const unsigned short* bCol = BT + (n0 + 16 * t + lcol) * (long)K + k0;
      c[t] = wmma_bf16(a, frag_b(bCol, half), c[t]);
    }
  }

#pragma unroll
  for (int t = 0; t < 4; ++t) {
#pragma unroll
    for (int j = 0; j < 8; ++j) {
      long row = m0 + j + 8 * half;
      long col = n0 + 16 * t + lcol;
      float v = c[t][j] + bias[col];
      if (relu) v = fmaxf(v, 0.f);
      if (rowscale) v *= rowscale[row * rs_stride + rs_off];
      if (resid) v += resid[row * (long)N + col];
      long idx = row * (long)N + col;
      if (Cf) { if (accum) Cf[idx] += v; else Cf[idx] = v; }
      if (Cb) {
        if (bf_mode == 0) {
          Cb[idx] = f2bf(v);
        } else {  // V^T scatter: token row = s*batchB + b -> [(b*N+col)*S + s]
          long s = row / batchB, bb = row % batchB;
          Cb[(bb * (long)N + col) * (long)(M / batchB) + s] = f2bf(v);
        }
      }
    }
  }
}

// ---------------------------------------------------------------------------
// Flash attention, wave32 WMMA. One wave = 16 queries x dh=64.
// Q,K bf16 row-major [T=S*B, 256] (head slice at h*64); V bf16 in
// [(b*256 + d)*Sk + s] layout so B-fragments are K-contiguous.
// Online softmax state per lane covers C-fragment row j + 8*half exactly.
// ---------------------------------------------------------------------------
__global__ __launch_bounds__(128) void flash_attn_wmma(
    const unsigned short* __restrict__ Qb, const unsigned short* __restrict__ Kb,
    const unsigned short* __restrict__ Vt, unsigned short* __restrict__ Ob,
    int Sq, int Sk, int Bb) {
  __shared__ __align__(16) unsigned short pbuf[4 * 16 * 32];
  const int lane = threadIdx.x & 31;
  const int wave = threadIdx.x >> 5;
  const int half = lane >> 4;
  const int lcol = lane & 15;
  const int b = blockIdx.y >> 2;      // H = 4
  const int h = blockIdx.y & 3;
  const long D = 256;
  const long m0 = (long)blockIdx.x * 64 + wave * 16;
  unsigned short* myp = pbuf + wave * 512;

  const unsigned short* qrow = Qb + ((m0 + lcol) * Bb + b) * D + h * 64;
  bfvec16 qa0 = frag_a(qrow, half);
  bfvec16 qa1 = frag_a(qrow + 32, half);

  float mrun[8], lrun[8];
  fvec8 acc[4] = {};
#pragma unroll
  for (int j = 0; j < 8; ++j) { mrun[j] = -1e30f; lrun[j] = 0.f; }

  const float scl = 0.125f;  // 1/sqrt(dh=64)
  for (int kc = 0; kc < Sk; kc += 32) {
    fvec8 s0 = {}, s1 = {};
    const unsigned short* kcol0 = Kb + ((long)(kc + lcol) * Bb + b) * D + h * 64;
    const unsigned short* kcol1 = Kb + ((long)(kc + 16 + lcol) * Bb + b) * D + h * 64;
    s0 = wmma_bf16(qa0, frag_b(kcol0, half), s0);
    s0 = wmma_bf16(qa1, frag_b(kcol0 + 32, half), s0);
    s1 = wmma_bf16(qa0, frag_b(kcol1, half), s1);
    s1 = wmma_bf16(qa1, frag_b(kcol1 + 32, half), s1);

#pragma unroll
    for (int j = 0; j < 8; ++j) {
      float a0 = s0[j] * scl, a1 = s1[j] * scl;
      float tm = fmaxf(a0, a1);
#pragma unroll
      for (int off = 8; off > 0; off >>= 1) tm = fmaxf(tm, __shfl_xor(tm, off, 32));
      float mn = fmaxf(mrun[j], tm);
      float corr = __expf(mrun[j] - mn);
      float p0 = __expf(a0 - mn), p1 = __expf(a1 - mn);
      float ts = p0 + p1;
#pragma unroll
      for (int off = 8; off > 0; off >>= 1) ts += __shfl_xor(ts, off, 32);
      lrun[j] = lrun[j] * corr + ts;
      mrun[j] = mn;
#pragma unroll
      for (int t = 0; t < 4; ++t) acc[t][j] *= corr;
      int prow = j + 8 * half;                 // C-layout row -> P row (query)
      myp[prow * 32 + lcol] = f2bf(p0);
      myp[prow * 32 + 16 + lcol] = f2bf(p1);
    }
    asm volatile("s_wait_dscnt 0" ::: "memory");  // LDS bounce: transpose C->A
    bfvec16 pa;
    {
      Frag16 f;
      const unsigned short* pp = myp + lcol * 32 + half * 8;
      f.u[0] = *(const uint4*)pp;
      f.u[1] = *(const uint4*)(pp + 16);
      pa = f.v;
    }
#pragma unroll
    for (int t = 0; t < 4; ++t) {
      const unsigned short* vcol =
          Vt + ((long)b * D + h * 64 + 16 * t + lcol) * (long)Sk + kc;
      acc[t] = wmma_bf16(pa, frag_b(vcol, half), acc[t]);
    }
    asm volatile("s_wait_dscnt 0" ::: "memory");
  }

  float inv[8];
#pragma unroll
  for (int j = 0; j < 8; ++j) inv[j] = 1.f / lrun[j];
#pragma unroll
  for (int t = 0; t < 4; ++t) {
#pragma unroll
    for (int j = 0; j < 8; ++j) {
      long tok = (m0 + j + 8 * half) * Bb + b;
      long col = h * 64 + 16 * t + lcol;
      Ob[tok * D + col] = f2bf(acc[t][j] * inv[j]);
    }
  }
}

// ---------------------------------------------------------------------------
// LayerNorm over D=256. One wave per row; optional residual, optional bf16 out.
// ---------------------------------------------------------------------------
__global__ __launch_bounds__(256) void layernorm_kernel(
    const float* __restrict__ X, const float* __restrict__ R,
    const float* __restrict__ g, const float* __restrict__ bta,
    float* __restrict__ Of, unsigned short* __restrict__ Ob, int Tn) {
  int lane = threadIdx.x & 31, wave = threadIdx.x >> 5;
  long row = (long)blockIdx.x * 8 + wave;
  if (row >= Tn) return;
  const float* xr = X + row * 256;
  const float* rr = R ? R + row * 256 : nullptr;
  float v[8], s = 0.f, sq = 0.f;
#pragma unroll
  for (int i = 0; i < 8; ++i) {
    int c = lane + i * 32;
    float x = xr[c] + (rr ? rr[c] : 0.f);
    v[i] = x; s += x; sq += x * x;
  }
#pragma unroll
  for (int off = 16; off > 0; off >>= 1) {
    s += __shfl_xor(s, off, 32);
    sq += __shfl_xor(sq, off, 32);
  }
  float mu = s * (1.f / 256.f);
  float var = sq * (1.f / 256.f) - mu * mu;
  float rs = rsqrtf(var + 1e-5f);
#pragma unroll
  for (int i = 0; i < 8; ++i) {
    int c = lane + i * 32;
    float y = (v[i] - mu) * rs * g[c] + bta[c];
    Of[row * 256 + c] = y;
    if (Ob) Ob[row * 256 + c] = f2bf(y);
  }
}

// ---------------------------------------------------------------------------
// MoE gating: softmax over E=8, top-2 renormalized comb weights, prob sums.
// ---------------------------------------------------------------------------
__global__ __launch_bounds__(256) void gate_kernel(
    const float* __restrict__ X, const float* __restrict__ Wg,
    const float* __restrict__ bg, float* __restrict__ comb,
    float* __restrict__ counts, int Tn) {
  long t = (long)blockIdx.x * blockDim.x + threadIdx.x;
  if (t >= Tn) return;
  float lg[8];
#pragma unroll
  for (int e = 0; e < 8; ++e) lg[e] = bg[e];
  const float* xr = X + t * 256;
  for (int d = 0; d < 256; ++d) {
    float x = xr[d];
    const float* w = Wg + d * 8;
#pragma unroll
    for (int e = 0; e < 8; ++e) lg[e] += x * w[e];
  }
  float mx = lg[0];
#pragma unroll
  for (int e = 1; e < 8; ++e) mx = fmaxf(mx, lg[e]);
  float p[8], den = 0.f;
#pragma unroll
  for (int e = 0; e < 8; ++e) { p[e] = __expf(lg[e] - mx); den += p[e]; }
  float invd = 1.f / den;
#pragma unroll
  for (int e = 0; e < 8; ++e) p[e] *= invd;
  int i0 = 0;
#pragma unroll
  for (int e = 1; e < 8; ++e) if (p[e] > p[i0]) i0 = e;
  int i1 = (i0 == 0) ? 1 : 0;
#pragma unroll
  for (int e = 0; e < 8; ++e) if (e != i0 && p[e] > p[i1]) i1 = e;
  float gs = p[i0] + p[i1];
#pragma unroll
  for (int e = 0; e < 8; ++e) comb[t * 8 + e] = 0.f;
  comb[t * 8 + i0] = p[i0] / gs;
  comb[t * 8 + i1] = p[i1] / gs;
#pragma unroll
  for (int e = 0; e < 8; ++e) atomicAdd(&counts[e], p[e]);
}

// aux = 0.01 * sum_parts( E * sum_e frac_e * mean_e ) = 0.01*sum 8*ss/(tot*T)
__global__ void aux_kernel(const float* __restrict__ counts,
                           float* __restrict__ out) {
  float aux = 0.f;
  for (int part = 0; part < 2; ++part) {
    float tot = 0.f, ss = 0.f;
    for (int e = 0; e < 8; ++e) {
      float c = counts[part * 8 + e];
      tot += c; ss += c * c;
    }
    aux += 8.f * ss / (tot * 8192.f);
  }
  out[0] = 0.01f * aux;
}

// ---------------------------------------------------------------------------
// Host orchestration
// ---------------------------------------------------------------------------
extern "C" void kernel_launch(void* const* d_in, const int* in_sizes, int n_in,
                              void* d_out, int out_size, void* d_ws,
                              size_t ws_size, hipStream_t stream) {
  (void)in_sizes; (void)n_in; (void)out_size; (void)ws_size;
  const long T = 8192, Dm = 256, F = 1024, S = 2048, Bb = 4;

  const float* x_context = (const float*)d_in[0];
  const float* x_target  = (const float*)d_in[1];
  const float* Wq = (const float*)d_in[2];
  const float* Wk = (const float*)d_in[3];
  const float* Wv = (const float*)d_in[4];
  const float* bq = (const float*)d_in[5];
  const float* bk = (const float*)d_in[6];
  const float* bv = (const float*)d_in[7];
  const float* Wo = (const float*)d_in[8];
  const float* bo = (const float*)d_in[9];
  const float* Wg = (const float*)d_in[10];
  const float* bg = (const float*)d_in[11];
  const float* W1 = (const float*)d_in[12];
  const float* b1 = (const float*)d_in[13];
  const float* W2 = (const float*)d_in[14];
  const float* b2 = (const float*)d_in[15];
  const float* g_c1 = (const float*)d_in[16];
  const float* b_c1 = (const float*)d_in[17];
  const float* g_c2 = (const float*)d_in[18];
  const float* b_c2 = (const float*)d_in[19];
  const float* g_t1 = (const float*)d_in[20];
  const float* b_t1 = (const float*)d_in[21];
  const float* g_t2 = (const float*)d_in[22];
  const float* b_t2 = (const float*)d_in[23];

  float* out = (float*)d_out;
  float* out_xc = out;
  float* out_xt = out + T * Dm;
  float* out_aux = out + 2 * T * Dm;

  char* ws = (char*)d_ws;
  size_t off = 0;
  auto allocp = [&](size_t bytes) -> char* {
    char* p = ws + off;
    off += (bytes + 255) & ~(size_t)255;
    return p;
  };
  unsigned short* WqT = (unsigned short*)allocp(Dm * Dm * 2);
  unsigned short* WkT = (unsigned short*)allocp(Dm * Dm * 2);
  unsigned short* WvT = (unsigned short*)allocp(Dm * Dm * 2);
  unsigned short* WoT = (unsigned short*)allocp(Dm * Dm * 2);
  unsigned short* W1T = (unsigned short*)allocp(8 * Dm * F * 2);
  unsigned short* W2T = (unsigned short*)allocp(8 * F * Dm * 2);
  unsigned short* a_bf    = (unsigned short*)allocp(T * Dm * 2);
  unsigned short* q_bf    = (unsigned short*)allocp(T * Dm * 2);
  unsigned short* k_bf    = (unsigned short*)allocp(T * Dm * 2);
  unsigned short* vT2     = (unsigned short*)allocp(Bb * Dm * S * 2);
  unsigned short* attn_bf = (unsigned short*)allocp(T * Dm * 2);
  float* tmp_f  = (float*)allocp(T * Dm * 4);
  float* x1_f   = (float*)allocp(T * Dm * 4);
  unsigned short* x1_bf  = (unsigned short*)allocp(T * Dm * 2);
  unsigned short* xc2_bf = (unsigned short*)allocp(T * Dm * 2);
  unsigned short* h_bf   = (unsigned short*)allocp(T * F * 2);
  float* moe_f  = (float*)allocp(T * Dm * 4);
  float* comb   = (float*)allocp(T * 8 * 4);
  float* counts = (float*)allocp(16 * 4);

  dim3 blk128(128), blk256(256);
  auto gemm = [&](const unsigned short* A, const unsigned short* BT,
                  const float* bias, float* Cf, unsigned short* Cb,
                  const float* resid, const float* rowscale, int rs_off, int M,
                  int N, int K, int relu, int accum, int bf_mode) {
    dim3 grid(M / 64, N / 64);
    gemm_bf16_wmma<<<grid, blk128, 0, stream>>>(A, BT, bias, Cf, Cb, resid,
                                                rowscale, 8, rs_off, M, N, K,
                                                relu, accum, bf_mode, (int)Bb);
  };
  auto cvt = [&](const float* in, unsigned short* o, long n) {
    cvt_bf16_kernel<<<(n + 255) / 256, blk256, 0, stream>>>(in, o, n);
  };
  auto cvtT = [&](const float* in, unsigned short* o, int Eb, int R, int C) {
    long n = (long)Eb * R * C;
    cvt_transpose_kernel<<<(n + 255) / 256, blk256, 0, stream>>>(in, o, Eb, R, C);
  };
  auto ln = [&](const float* X, const float* R, const float* g, const float* b,
                float* Of, unsigned short* Ob) {
    layernorm_kernel<<<T / 8, blk256, 0, stream>>>(X, R, g, b, Of, Ob, (int)T);
  };

  // ---- weight prep (per-launch, deterministic) ----
  cvtT(Wq, WqT, 1, Dm, Dm);
  cvtT(Wk, WkT, 1, Dm, Dm);
  cvtT(Wv, WvT, 1, Dm, Dm);
  cvtT(Wo, WoT, 1, Dm, Dm);
  cvtT(W1, W1T, 8, Dm, F);   // -> per-expert [F, Dm]
  cvtT(W2, W2T, 8, F, Dm);   // -> per-expert [Dm, F]
  zero_kernel<<<1, 32, 0, stream>>>(counts, 16);

  dim3 attn_grid(S / 64, Bb * 4);

  // ================= context: self-attention + MoE =================
  cvt(x_context, a_bf, T * Dm);
  gemm(a_bf, WqT, bq, nullptr, q_bf, nullptr, nullptr, 0, T, Dm, Dm, 0, 0, 0);
  gemm(a_bf, WkT, bk, nullptr, k_bf, nullptr, nullptr, 0, T, Dm, Dm, 0, 0, 0);
  gemm(a_bf, WvT, bv, nullptr, vT2, nullptr, nullptr, 0, T, Dm, Dm, 0, 0, 1);
  flash_attn_wmma<<<attn_grid, blk128, 0, stream>>>(q_bf, k_bf, vT2, attn_bf,
                                                    (int)S, (int)S, (int)Bb);
  gemm(attn_bf, WoT, bo, tmp_f, nullptr, x_context, nullptr, 0, T, Dm, Dm, 0, 0, 0);
  ln(tmp_f, nullptr, g_c1, b_c1, x1_f, x1_bf);
  gate_kernel<<<T / 256, blk256, 0, stream>>>(x1_f, Wg, bg, comb, counts, (int)T);
  for (int e = 0; e < 8; ++e) {
    gemm(x1_bf, W1T + (long)e * F * Dm, b1 + (long)e * F, nullptr, h_bf, nullptr,
         nullptr, 0, T, F, Dm, 1, 0, 0);
    gemm(h_bf, W2T + (long)e * Dm * F, b2 + (long)e * Dm, moe_f, nullptr, nullptr,
         comb, e, T, Dm, F, 0, (e > 0), 0);
  }
  ln(moe_f, x1_f, g_c2, b_c2, out_xc, xc2_bf);

  // ================= target: cross-attention + MoE =================
  cvt(x_target, a_bf, T * Dm);
  gemm(a_bf, WqT, bq, nullptr, q_bf, nullptr, nullptr, 0, T, Dm, Dm, 0, 0, 0);
  gemm(xc2_bf, WkT, bk, nullptr, k_bf, nullptr, nullptr, 0, T, Dm, Dm, 0, 0, 0);
  gemm(xc2_bf, WvT, bv, nullptr, vT2, nullptr, nullptr, 0, T, Dm, Dm, 0, 0, 1);
  flash_attn_wmma<<<attn_grid, blk128, 0, stream>>>(q_bf, k_bf, vT2, attn_bf,
                                                    (int)S, (int)S, (int)Bb);
  gemm(attn_bf, WoT, bo, tmp_f, nullptr, x_target, nullptr, 0, T, Dm, Dm, 0, 0, 0);
  ln(tmp_f, nullptr, g_t1, b_t1, x1_f, x1_bf);
  gate_kernel<<<T / 256, blk256, 0, stream>>>(x1_f, Wg, bg, comb, counts + 8, (int)T);
  for (int e = 0; e < 8; ++e) {
    gemm(x1_bf, W1T + (long)e * F * Dm, b1 + (long)e * F, nullptr, h_bf, nullptr,
         nullptr, 0, T, F, Dm, 1, 0, 0);
    gemm(h_bf, W2T + (long)e * Dm * F, b2 + (long)e * Dm, moe_f, nullptr, nullptr,
         comb, e, T, Dm, F, 0, (e > 0), 0);
  }
  ln(moe_f, x1_f, g_t2, b_t2, out_xt, nullptr);

  aux_kernel<<<1, 1, 0, stream>>>(counts, out_aux);
}
